// TransformerEncoder_60378650247492
// MI455X (gfx1250) — matrix operand order repaired and checked by reading, not compiled, not run on hardware
//
#include <hip/hip_runtime.h>
#include <hip/hip_bf16.h>

#define BB 4
#define LL 1024
#define DD 768
#define HH 12
#define FF 3072
#define DKK 64
#define PP 32
#define MAXD 256
#define MM (BB*LL)   // 4096 rows

typedef __attribute__((ext_vector_type(16))) __bf16 v16bf;
typedef __attribute__((ext_vector_type(8)))  float  v8f;
typedef __attribute__((ext_vector_type(4)))  unsigned v4u;
typedef __attribute__((ext_vector_type(8)))  unsigned v8u;

union Frag { v16bf v; uint4 q[2]; };

static __device__ __forceinline__ unsigned short f2bf(float f) {
    union { float f; unsigned u; } x; x.f = f;
    unsigned r = x.u + 0x7FFFu + ((x.u >> 16) & 1u);
    return (unsigned short)(r >> 16);
}

static __device__ __forceinline__ v8f wmma_bf16(v16bf a, v16bf b, v8f c) {
    return __builtin_amdgcn_wmma_f32_16x16x32_bf16(
        /*neg_a=*/false, a, /*neg_b=*/false, b,
        /*c_mod=*/(short)0, c, /*reuse_a=*/false, /*reuse_b=*/false);
}

// ---- TDM: DMA a [64 rows x 32 cols] bf16 tile from row-major [N,K] into LDS ----
// D# per CDNA5 ISA. Uniform (SGPR) operands; tracked by TENSORcnt.
static __device__ __forceinline__ void tdm_load_b_tile(const unsigned short* gsrc,
                                                       unsigned lds_off, int K, int N) {
    unsigned long long ga = (unsigned long long)(uintptr_t)gsrc;
    v4u g0;
    g0[0] = 1u;                                                   // count=1, user mode
    g0[1] = lds_off;                                              // lds_addr (bytes)
    g0[2] = (unsigned)ga;                                         // global_addr[31:0]
    g0[3] = (unsigned)((ga >> 32) & 0x1FFFFFFull) | (2u << 30);   // addr[56:32] | type=2
    v8u g1;
    g1[0] = 1u << 16;                                             // data_size=1 (2 bytes)
    g1[1] = ((unsigned)K & 0xFFFFu) << 16;                        // tensor_dim0[15:0]
    g1[2] = (((unsigned)K >> 16) & 0xFFFFu)                       // tensor_dim0[31:16]
          | (((unsigned)N & 0xFFFFu) << 16);                      // tensor_dim1[15:0]
    g1[3] = (((unsigned)N >> 16) & 0xFFFFu)                       // tensor_dim1[31:16]
          | (32u << 16);                                          // tile_dim0 = 32
    g1[4] = 64u;                                                  // tile_dim1=64, tile_dim2=0
    g1[5] = (unsigned)K;                                          // tensor_dim0_stride[31:0]
    g1[6] = 0u;                                                   // stride hi, dim1_stride lo
    g1[7] = 0u;
    v4u gz = {0u, 0u, 0u, 0u};
    asm volatile("tensor_load_to_lds %0, %1, %2, %3"
                 :: "s"(g0), "s"(g1), "s"(gz), "s"(gz)
                 : "memory");
}

// ---------------- weight fp32 [K,N] -> bf16 W^T [N,K] ----------------
__global__ __launch_bounds__(256)
void wt_bf16(const float* __restrict__ W, unsigned short* __restrict__ Wt, int K, int N) {
    int idx = blockIdx.x * 256 + threadIdx.x;
    if (idx >= K * N) return;
    int k = idx / N, n = idx % N;
    Wt[(size_t)n * K + k] = f2bf(W[idx]);
}

// ---------------- LayerNorm (D=768) fp32 in -> bf16 out ----------------
__global__ __launch_bounds__(256)
void ln_bf16(const float* __restrict__ in, const float* __restrict__ g,
             const float* __restrict__ bta, unsigned short* __restrict__ out, float eps) {
    int row = blockIdx.x, tid = threadIdx.x;
    const float* xr = in + (size_t)row * DD;
    float v0 = xr[tid], v1 = xr[tid + 256], v2 = xr[tid + 512];
    __shared__ float red[256];
    red[tid] = v0 + v1 + v2;
    __syncthreads();
    for (int o = 128; o > 0; o >>= 1) { if (tid < o) red[tid] += red[tid + o]; __syncthreads(); }
    float mu = red[0] * (1.f / DD);
    __syncthreads();
    float d0 = v0 - mu, d1 = v1 - mu, d2 = v2 - mu;
    red[tid] = d0 * d0 + d1 * d1 + d2 * d2;
    __syncthreads();
    for (int o = 128; o > 0; o >>= 1) { if (tid < o) red[tid] += red[tid + o]; __syncthreads(); }
    float rs = rsqrtf(red[0] * (1.f / DD) + eps);
    unsigned short* orow = out + (size_t)row * DD;
    orow[tid]       = f2bf(d0 * rs * g[tid]       + bta[tid]);
    orow[tid + 256] = f2bf(d1 * rs * g[tid + 256] + bta[tid + 256]);
    orow[tid + 512] = f2bf(d2 * rs * g[tid + 512] + bta[tid + 512]);
}

// ---------------- generic bf16 WMMA GEMM: C[M,N] = A[M,K] * Bt[N,K]^T ----------------
// B tile staged by the Tensor Data Mover, double-buffered (one barrier / K-step).
// epi: 0 = fp32 out + bias + optional residual
//      1 = bf16 [B,H,L,dk] layout, (acc+bias)*scale      (Q with 1/8, K with 1.0)
//      2 = bf16 [B,H,dk,L] layout (V transposed)
//      3 = bf16 row-major, exact GELU(acc+bias)
__global__ __launch_bounds__(256)
void gemm_bf16(const unsigned short* __restrict__ A, const unsigned short* __restrict__ Bt,
               const float* __restrict__ bias, const float* __restrict__ res,
               float* __restrict__ outF, unsigned short* __restrict__ outB,
               int N, int K, int epi, float scale) {
    const int wave = threadIdx.x >> 5, lane = threadIdx.x & 31;
    const int half = lane >> 4, lm = lane & 15;
    const int m0 = blockIdx.y * 128 + wave * 16;
    const int n0 = blockIdx.x * 64;

    __shared__ __align__(16) unsigned short bs[2][64 * 32];   // 2 x 4KB B tiles
    unsigned lds0 = (unsigned)(unsigned long long)
        (__attribute__((address_space(3))) unsigned short*)&bs[0][0];

    v8f acc[4];
    const v8f zf = {0.f, 0.f, 0.f, 0.f, 0.f, 0.f, 0.f, 0.f};
#pragma unroll
    for (int t = 0; t < 4; t++) acc[t] = zf;

    // prime the pipeline: DMA tile for k0=0 into buffer 0
    if (wave == 0)
        tdm_load_b_tile(&Bt[(size_t)n0 * K], lds0, K, N);

    int buf = 0;
    for (int k0 = 0; k0 < K; k0 += 32) {
        if (wave == 0) __builtin_amdgcn_s_wait_tensorcnt(0);
        __syncthreads();   // tile(k0) visible to all; everyone done with other buffer
        if (wave == 0 && (k0 + 32) < K)
            tdm_load_b_tile(&Bt[(size_t)n0 * K + k0 + 32],
                            lds0 + (unsigned)((buf ^ 1) * 4096), K, N);

        const unsigned short* ap = &A[(size_t)(m0 + lm) * K + k0 + 8 * half];
        __builtin_prefetch(ap + 64, 0, 1);   // cover next A tiles
        Frag a;
        a.q[0] = *(const uint4*)ap;
        a.q[1] = *(const uint4*)(ap + 16);

        const unsigned short* bsb = bs[buf];
#pragma unroll
        for (int t = 0; t < 4; t++) {
            Frag bfr;
            const unsigned short* bp = &bsb[(t * 16 + lm) * 32 + 8 * half];
            bfr.q[0] = *(const uint4*)bp;
            bfr.q[1] = *(const uint4*)(bp + 16);
            acc[t] = wmma_bf16(a.v, bfr.v, acc[t]);
        }
        buf ^= 1;
    }

#pragma unroll
    for (int t = 0; t < 4; t++) {
#pragma unroll
        for (int v = 0; v < 8; v++) {
            int row = m0 + v + 8 * half;
            int col = n0 + t * 16 + lm;
            float x = acc[t][v] + bias[col];
            if (epi == 0) {
                size_t i = (size_t)row * N + col;
                outF[i] = x + (res ? res[i] : 0.f);
            } else if (epi == 1) {
                int b = row >> 10, l = row & 1023, h = col >> 6, dk = col & 63;
                outB[(((size_t)(b * HH + h)) * LL + l) * DKK + dk] = f2bf(x * scale);
            } else if (epi == 2) {
                int b = row >> 10, l = row & 1023, h = col >> 6, dk = col & 63;
                outB[(((size_t)(b * HH + h)) * DKK + dk) * LL + l] = f2bf(x);
            } else {
                float gx = 0.5f * x * (1.0f + erff(x * 0.70710678118654752f));
                outB[(size_t)row * N + col] = f2bf(gx);
            }
        }
    }
}

// ---------------- flash attention, one wave per 16-query tile ----------------
__global__ __launch_bounds__(32)
void attn_wmma(const unsigned short* __restrict__ Q,   // [B,H,L,dk] bf16, pre-scaled 1/8
               const unsigned short* __restrict__ Kb,  // [B,H,L,dk] bf16
               const unsigned short* __restrict__ Vt,  // [B,H,dk,L] bf16
               const int* __restrict__ postag, const float* __restrict__ lex,
               const float* __restrict__ rel_emb, const float* __restrict__ ptable,
               unsigned short* __restrict__ attnOut)   // [B*L, D] bf16
{
    const int qt = blockIdx.x, h = blockIdx.y, b = blockIdx.z;
    const int lane = threadIdx.x, half = lane >> 4, lm = lane & 15;
    const size_t bh = (size_t)(b * HH + h);
    const unsigned short* Qp = Q + (bh * LL + qt * 16) * DKK;
    const unsigned short* Kp = Kb + bh * LL * DKK;
    const unsigned short* Vp = Vt + bh * DKK * LL;

    Frag aQ[2];
    {
        const unsigned short* qr = Qp + lm * DKK;
        aQ[0].q[0] = *(const uint4*)(qr + 8 * half);
        aQ[0].q[1] = *(const uint4*)(qr + 16 + 8 * half);
        aQ[1].q[0] = *(const uint4*)(qr + 32 + 8 * half);
        aQ[1].q[1] = *(const uint4*)(qr + 48 + 8 * half);
    }

    int   ptm[8];
    float rmax[8], rsum[8];
#pragma unroll
    for (int v = 0; v < 8; v++) {
        int row = qt * 16 + v + 8 * half;
        ptm[v]  = postag[b * LL + row] * PP;
        rmax[v] = -1e30f;
        rsum[v] = 0.f;
    }
    const v8f zf = {0.f, 0.f, 0.f, 0.f, 0.f, 0.f, 0.f, 0.f};
    v8f O[4];
#pragma unroll
    for (int d = 0; d < 4; d++) O[d] = zf;

    __shared__ __align__(16) unsigned short pS[16 * 32];   // 1KB P staging

    for (int s0 = 0; s0 < LL; s0 += 32) {
        if (s0 + 32 < LL) {  // prefetch next K/V tiles
            __builtin_prefetch(Kp + (size_t)(s0 + 32 + lm) * DKK, 0, 1);
            __builtin_prefetch(Vp + (size_t)lm * LL + s0 + 32, 0, 1);
        }
        v8f lg[2]; lg[0] = zf; lg[1] = zf;
#pragma unroll
        for (int t = 0; t < 2; t++) {
            const unsigned short* kr = Kp + (size_t)(s0 + t * 16 + lm) * DKK;
            Frag b0, b1;
            b0.q[0] = *(const uint4*)(kr + 8 * half);
            b0.q[1] = *(const uint4*)(kr + 16 + 8 * half);
            b1.q[0] = *(const uint4*)(kr + 32 + 8 * half);
            b1.q[1] = *(const uint4*)(kr + 48 + 8 * half);
            lg[t] = wmma_bf16(aQ[0].v, b0.v, lg[t]);
            lg[t] = wmma_bf16(aQ[1].v, b1.v, lg[t]);
        }
        int pts[2]; float lx[2];
#pragma unroll
        for (int t = 0; t < 2; t++) {
            int s = s0 + t * 16 + lm;
            pts[t] = postag[b * LL + s];
            lx[t]  = lex[b * LL + s];
        }
#pragma unroll
        for (int v = 0; v < 8; v++) {
            int row = qt * 16 + v + 8 * half;
#pragma unroll
            for (int t = 0; t < 2; t++) {
                int s = s0 + t * 16 + lm;
                int rel = s - row;
                rel = rel > MAXD ? MAXD : (rel < -MAXD ? -MAXD : rel);
                float bias = rel_emb[(rel + MAXD) * HH + h]
                           + ptable[(ptm[v] + pts[t]) * HH + h]
                           + lx[t];
                lg[t][v] += bias;
            }
        }
        // online softmax per row (row lives in 16 lanes of one half: xor 1/2/4/8)
#pragma unroll
        for (int v = 0; v < 8; v++) {
            float vmax = fmaxf(lg[0][v], lg[1][v]);
#pragma unroll
            for (int off = 1; off < 16; off <<= 1)
                vmax = fmaxf(vmax, __shfl_xor(vmax, off, 32));
            float newm = fmaxf(rmax[v], vmax);
            float f  = __expf(rmax[v] - newm);
            float p0 = __expf(lg[0][v] - newm);
            float p1 = __expf(lg[1][v] - newm);
            float ps = p0 + p1;
#pragma unroll
            for (int off = 1; off < 16; off <<= 1)
                ps += __shfl_xor(ps, off, 32);
            rsum[v] = rsum[v] * f + ps;
            rmax[v] = newm;
#pragma unroll
            for (int d = 0; d < 4; d++) O[d][v] *= f;
            pS[(v + 8 * half) * 32 + lm]      = f2bf(p0);
            pS[(v + 8 * half) * 32 + 16 + lm] = f2bf(p1);
        }
        __syncthreads();
        Frag aP;
        const unsigned short* pr = &pS[lm * 32 + 8 * half];
        aP.q[0] = *(const uint4*)pr;
        aP.q[1] = *(const uint4*)(pr + 16);
#pragma unroll
        for (int d = 0; d < 4; d++) {
            const unsigned short* vr = Vp + (size_t)(d * 16 + lm) * LL + s0 + 8 * half;
            Frag bV;
            bV.q[0] = *(const uint4*)vr;
            bV.q[1] = *(const uint4*)(vr + 16);
            O[d] = wmma_bf16(aP.v, bV.v, O[d]);
        }
        __syncthreads();
    }

#pragma unroll
    for (int v = 0; v < 8; v++) {
        float inv = 1.0f / rsum[v];
        int row = qt * 16 + v + 8 * half;
#pragma unroll
        for (int d = 0; d < 4; d++)
            attnOut[(size_t)(b * LL + row) * DD + h * DKK + d * 16 + lm] = f2bf(O[d][v] * inv);
    }
}

extern "C" void kernel_launch(void* const* d_in, const int* in_sizes, int n_in,
                              void* d_out, int out_size, void* d_ws, size_t ws_size,
                              hipStream_t stream) {
    const float* x     = (const float*)d_in[0];
    const int*   ptag  = (const int*)  d_in[1];
    const float* lex   = (const float*)d_in[2];
    const float* ln1g  = (const float*)d_in[3];
    const float* ln1b  = (const float*)d_in[4];
    const float* Wq    = (const float*)d_in[5];
    const float* bq    = (const float*)d_in[6];
    const float* Wk    = (const float*)d_in[7];
    const float* bk    = (const float*)d_in[8];
    const float* Wv    = (const float*)d_in[9];
    const float* bv    = (const float*)d_in[10];
    const float* Wo    = (const float*)d_in[11];
    const float* bo    = (const float*)d_in[12];
    const float* rel   = (const float*)d_in[13];
    const float* ptab  = (const float*)d_in[14];
    const float* ln2g  = (const float*)d_in[15];
    const float* ln2b  = (const float*)d_in[16];
    const float* W1    = (const float*)d_in[17];
    const float* b1    = (const float*)d_in[18];
    const float* W2    = (const float*)d_in[19];
    const float* b2    = (const float*)d_in[20];
    float* out         = (float*)d_out;

    // workspace carve-up (256B aligned)
    unsigned char* w = (unsigned char*)d_ws;
    size_t off = 0;
    auto alloc = [&](size_t bytes) {
        size_t o = off;
        off = (off + bytes + 255) & ~(size_t)255;
        return (void*)(w + o);
    };
    unsigned short* xln  = (unsigned short*)alloc((size_t)MM * DD * 2);
    unsigned short* WqT  = (unsigned short*)alloc((size_t)DD * DD * 2);
    unsigned short* WkT  = (unsigned short*)alloc((size_t)DD * DD * 2);
    unsigned short* WvT  = (unsigned short*)alloc((size_t)DD * DD * 2);
    unsigned short* WoT  = (unsigned short*)alloc((size_t)DD * DD * 2);
    unsigned short* W1T  = (unsigned short*)alloc((size_t)DD * FF * 2);
    unsigned short* W2T  = (unsigned short*)alloc((size_t)FF * DD * 2);
    unsigned short* Qb   = (unsigned short*)alloc((size_t)MM * DD * 2);
    unsigned short* Kbuf = (unsigned short*)alloc((size_t)MM * DD * 2);
    unsigned short* Vtb  = (unsigned short*)alloc((size_t)MM * DD * 2);
    unsigned short* attn = (unsigned short*)alloc((size_t)MM * DD * 2);
    float*          outb = (float*)        alloc((size_t)MM * DD * 4);
    unsigned short* hln  = (unsigned short*)alloc((size_t)MM * DD * 2);
    unsigned short* hff  = (unsigned short*)alloc((size_t)MM * FF * 2);
    (void)ws_size; (void)n_in; (void)in_sizes; (void)out_size;

    // 1) weight convert+transpose to bf16
    {
        int nDD = (DD * DD + 255) / 256, nDF = (DD * FF + 255) / 256;
        wt_bf16<<<nDD, 256, 0, stream>>>(Wq, WqT, DD, DD);
        wt_bf16<<<nDD, 256, 0, stream>>>(Wk, WkT, DD, DD);
        wt_bf16<<<nDD, 256, 0, stream>>>(Wv, WvT, DD, DD);
        wt_bf16<<<nDD, 256, 0, stream>>>(Wo, WoT, DD, DD);
        wt_bf16<<<nDF, 256, 0, stream>>>(W1, W1T, DD, FF);
        wt_bf16<<<nDF, 256, 0, stream>>>(W2, W2T, FF, DD);
    }
    // 2) LN1
    ln_bf16<<<MM, 256, 0, stream>>>(x, ln1g, ln1b, xln, 1e-6f);

    // 3) QKV projections (fold softmax 1/sqrt(64)=0.125 into Q)
    dim3 gD(DD / 64, MM / 128);
    gemm_bf16<<<gD, 256, 0, stream>>>(xln, WqT, bq, nullptr, nullptr, Qb,   DD, DD, 1, 0.125f);
    gemm_bf16<<<gD, 256, 0, stream>>>(xln, WkT, bk, nullptr, nullptr, Kbuf, DD, DD, 1, 1.0f);
    gemm_bf16<<<gD, 256, 0, stream>>>(xln, WvT, bv, nullptr, nullptr, Vtb,  DD, DD, 2, 1.0f);

    // 4) fused flash attention with rel-pos + postag + lex biases
    dim3 gA(LL / 16, HH, BB);
    attn_wmma<<<gA, 32, 0, stream>>>(Qb, Kbuf, Vtb, ptag, lex, rel, ptab, attn);

    // 5) output projection + residual (fp32)
    gemm_bf16<<<gD, 256, 0, stream>>>(attn, WoT, bo, x, outb, nullptr, DD, DD, 0, 1.0f);

    // 6) LN2 -> bf16
    ln_bf16<<<MM, 256, 0, stream>>>(outb, ln2g, ln2b, hln, 1e-5f);

    // 7) FFN up + exact GELU -> bf16
    dim3 gF(FF / 64, MM / 128);
    gemm_bf16<<<gF, 256, 0, stream>>>(hln, W1T, b1, nullptr, nullptr, hff, FF, DD, 3, 1.0f);

    // 8) FFN down + residual -> fp32 d_out
    gemm_bf16<<<gD, 256, 0, stream>>>(hff, W2T, b2, outb, out, nullptr, DD, FF, 0, 1.0f);
}
